// GATModel_66932770341399
// MI455X (gfx1250) — matrix operand order repaired and checked by reading, not compiled, not run on hardware
//
#include <hip/hip_runtime.h>
#include <hip/hip_bf16.h>

#define HCH 64        // hidden channels
#define HEADS 2
#define CHF (HEADS*HCH)   // 128 flattened edge channels
#define SLOPE 0.2f
#define NGRAPH 64
#define OUTC 10
#define BN_EPS 1e-5f

typedef __attribute__((ext_vector_type(16))) _Float16 v16h;
typedef __attribute__((ext_vector_type(8)))  float    v8f;

// ---------- ordered-uint mapping for float atomic max ----------
__device__ __forceinline__ unsigned f2ord(float f) {
  unsigned u = __float_as_uint(f);
  return (u & 0x80000000u) ? ~u : (u | 0x80000000u);
}
__device__ __forceinline__ float ord2f(unsigned u) {
  return (u & 0x80000000u) ? __uint_as_float(u & 0x7fffffffu)
                           : __uint_as_float(~u);
}

// ---------- WMMA fragment loads (ISA 7.12.2 16-bit layouts) ----------
// A 16x32 f16: lane m%16 (dup over lane/16=g), per-lane K = two contiguous
// 8-float runs at k0+8g and k0+8g+16  ->  4x float4 loads.
__device__ __forceinline__ v16h load_a_frag(const float* __restrict__ X,
                                            int m0, int k0, int nrow) {
  const int lane = threadIdx.x & 31;
  int m = m0 + (lane & 15);
  m = m < nrow ? m : nrow - 1;            // branchless clamp: EXEC stays all-1s
  const int g = lane >> 4;
  const float* p0 = X + (size_t)m * 64 + k0 + 8 * g;
  const float4 f0 = *(const float4*)(p0 + 0);
  const float4 f1 = *(const float4*)(p0 + 4);
  const float4 f2 = *(const float4*)(p0 + 16);
  const float4 f3 = *(const float4*)(p0 + 20);
  v16h a;
  a[0]  = (_Float16)f0.x; a[1]  = (_Float16)f0.y;
  a[2]  = (_Float16)f0.z; a[3]  = (_Float16)f0.w;
  a[4]  = (_Float16)f1.x; a[5]  = (_Float16)f1.y;
  a[6]  = (_Float16)f1.z; a[7]  = (_Float16)f1.w;
  a[8]  = (_Float16)f2.x; a[9]  = (_Float16)f2.y;
  a[10] = (_Float16)f2.z; a[11] = (_Float16)f2.w;
  a[12] = (_Float16)f3.x; a[13] = (_Float16)f3.y;
  a[14] = (_Float16)f3.z; a[15] = (_Float16)f3.w;
  return a;
}

// B 32x16 f16: N=lane%16, element i holds K=8g+i (i<8) or K=16+8g+(i-8) (i>=8).
template <int KO>
__device__ __forceinline__ v16h load_b_frag(const float* __restrict__ W,
                                            int k0, int n0) {
  const int lane = threadIdx.x & 31;
  const int n = n0 + (lane & 15);
  const int g = lane >> 4;
  const float* wp = W + (size_t)(k0 + 8 * g) * KO + n;
  v16h b;
#pragma unroll
  for (int i = 0; i < 16; ++i)
    b[i] = (_Float16)wp[(size_t)(i + (i & 8)) * KO];
  return b;
}

// Y[nrow,KO] = X[nrow,64] @ W[64,KO] + bias.  One wave = one 16x16 tile.
// blockDim.x = 32 * (KO/16); blockIdx.x = row tile.
template <int KO>
__global__ void gemm_wmma_kernel(const float* __restrict__ X,
                                 const float* __restrict__ W,
                                 const float* __restrict__ bias,
                                 float* __restrict__ Y, int nrow) {
  const int wave = threadIdx.x >> 5;
  const int lane = threadIdx.x & 31;
  const int m0 = blockIdx.x << 4;
  const int n0 = wave << 4;
  v8f c = {};
  {
    v16h a = load_a_frag(X, m0, 0, nrow);
    v16h b = load_b_frag<KO>(W, 0, n0);
    c = __builtin_amdgcn_wmma_f32_16x16x32_f16(false, a, false, b,
                                               (short)0, c, false, false);
  }
  {
    v16h a = load_a_frag(X, m0, 32, nrow);
    v16h b = load_b_frag<KO>(W, 32, n0);
    c = __builtin_amdgcn_wmma_f32_16x16x32_f16(false, a, false, b,
                                               (short)0, c, false, false);
  }
  const int col = n0 + (lane & 15);
  const float bv = bias[col];
  const int rbase = m0 + ((lane >> 4) << 3);
  float* __restrict__ yp = Y + (size_t)rbase * KO + col;
  if (rbase + 8 <= nrow) {          // full tile: constant-offset stores
#pragma unroll
    for (int r = 0; r < 8; ++r) yp[r * KO] = c[r] + bv;
  } else {
#pragma unroll
    for (int r = 0; r < 8; ++r)
      if (rbase + r < nrow) yp[r * KO] = c[r] + bv;
  }
}

// ---------- edge helpers ----------
__device__ __forceinline__ void edge_nodes(const int* __restrict__ ei, int E,
                                           int e, int& src, int& dst) {
  if (e < E) { src = ei[e]; dst = ei[E + e]; }
  else       { src = dst = e - E; }          // appended self loops
}

// per-layer init: hnew = bias (broadcast), m = ord(-inf)=0, den = 0, stats = 0
__global__ void layer_init_kernel(float* __restrict__ hnew,
                                  const float* __restrict__ bias,
                                  unsigned* __restrict__ mOrd,
                                  float* __restrict__ den,
                                  float* __restrict__ stats, int N) {
  const int tid = blockIdx.x * blockDim.x + threadIdx.x;
  if (tid < N * HCH) hnew[tid] = bias[tid & (HCH - 1)];
  if (tid < N * HEADS) { mOrd[tid] = 0u; den[tid] = 0.f; }
  if (tid < 256) stats[tid] = 0.f;
}

// e_score = att . leaky_relu(xl[src]+xr[dst]); wave per edge, 16 lanes per head
__global__ void edge_score_kernel(const float* __restrict__ xl,
                                  const float* __restrict__ xr,
                                  const float* __restrict__ att,
                                  const int* __restrict__ ei,
                                  float* __restrict__ esc,
                                  unsigned* __restrict__ mOrd, int E, int Ep) {
  const int e = blockIdx.x * (blockDim.x >> 5) + (threadIdx.x >> 5);
  if (e >= Ep) return;
  const int lane = threadIdx.x & 31;
  int src, dst; edge_nodes(ei, E, e, src, dst);
  const int c0 = lane << 2;
  const float4 lv = *(const float4*)(xl + (size_t)src * CHF + c0);
  const float4 rv = *(const float4*)(xr + (size_t)dst * CHF + c0);
  const float4 av = *(const float4*)(att + c0);
  float p = 0.f, z;
  z = lv.x + rv.x; z = z > 0.f ? z : SLOPE * z; p += z * av.x;
  z = lv.y + rv.y; z = z > 0.f ? z : SLOPE * z; p += z * av.y;
  z = lv.z + rv.z; z = z > 0.f ? z : SLOPE * z; p += z * av.z;
  z = lv.w + rv.w; z = z > 0.f ? z : SLOPE * z; p += z * av.w;
#pragma unroll
  for (int off = 1; off < 16; off <<= 1) p += __shfl_xor(p, off, 32);
  if ((lane & 15) == 0) {
    const int head = lane >> 4;
    esc[(size_t)e * HEADS + head] = p;
    atomicMax(&mOrd[dst * HEADS + head], f2ord(p));
  }
}

// ex = exp(e - m[dst]); den[dst] += ex   (one thread per edge*head)
__global__ void edge_exp_kernel(const int* __restrict__ ei,
                                const unsigned* __restrict__ mOrd,
                                float* __restrict__ esc,
                                float* __restrict__ den, int E, int Ep) {
  const int tid = blockIdx.x * blockDim.x + threadIdx.x;
  if (tid >= HEADS * Ep) return;
  const int e = tid >> 1, hd = tid & 1;
  int src, dst; edge_nodes(ei, E, e, src, dst);
  const float mx = ord2f(mOrd[dst * HEADS + hd]);
  const float ex = __expf(esc[tid] - mx);
  esc[tid] = ex;
  atomicAdd(&den[dst * HEADS + hd], ex);
}

// hnew[dst,c] += 0.5 * alpha * xl[src,h,c]  (mean over heads fused)
__global__ void edge_aggr_kernel(const float* __restrict__ xl,
                                 const int* __restrict__ ei,
                                 const float* __restrict__ esc,
                                 const float* __restrict__ den,
                                 float* __restrict__ hnew, int E, int Ep) {
  const int e = blockIdx.x * (blockDim.x >> 5) + (threadIdx.x >> 5);
  if (e >= Ep) return;
  const int lane = threadIdx.x & 31;
  int src, dst; edge_nodes(ei, E, e, src, dst);
  const int head = lane >> 4;
  const float alpha = esc[(size_t)e * HEADS + head] / den[dst * HEADS + head];
  const int c0 = lane << 2;
  const float4 lv = *(const float4*)(xl + (size_t)src * CHF + c0);
  const float s = 0.5f * alpha;
  float* dp = hnew + (size_t)dst * HCH + (c0 & (HCH - 1));
  atomicAdd(dp + 0, s * lv.x);
  atomicAdd(dp + 1, s * lv.y);
  atomicAdd(dp + 2, s * lv.z);
  atomicAdd(dp + 3, s * lv.w);
}

// per-channel sum / sumsq with block reduction; blockDim = (64,4), 1024 rows/block
__global__ void bn_stats_kernel(const float* __restrict__ hnew,
                                float* __restrict__ stats, int N) {
  __shared__ float ssum[4][HCH];
  __shared__ float ssq[4][HCH];
  const int c = threadIdx.x, ty = threadIdx.y;
  const int r0 = blockIdx.x * 1024;
  const int rEnd = (r0 + 1024 < N) ? r0 + 1024 : N;
  float s = 0.f, q = 0.f;
  for (int r = r0 + ty; r < rEnd; r += 4) {
    const float v = hnew[(size_t)r * HCH + c];
    s += v; q += v * v;
  }
  ssum[ty][c] = s; ssq[ty][c] = q;
  __syncthreads();
  if (ty == 0) {
    s = ssum[0][c] + ssum[1][c] + ssum[2][c] + ssum[3][c];
    q = ssq[0][c] + ssq[1][c] + ssq[2][c] + ssq[3][c];
    atomicAdd(&stats[c], s);
    atomicAdd(&stats[HCH + c], q);
  }
}

// fold mean/var/gamma/beta into per-channel scale+shift
__global__ void bn_finalize_kernel(float* __restrict__ stats,
                                   const float* __restrict__ gamma,
                                   const float* __restrict__ beta, int N) {
  const int c = threadIdx.x;
  if (c >= HCH) return;
  const float inv_n = 1.f / (float)N;
  const float mu = stats[c] * inv_n;
  const float var = stats[HCH + c] * inv_n - mu * mu;
  const float sc = gamma[c] * rsqrtf(var + BN_EPS);
  stats[128 + c] = sc;
  stats[192 + c] = beta[c] - mu * sc;
}

// h = relu(hnew * scale + shift)
__global__ void bn_apply_kernel(const float* __restrict__ hnew,
                                const float* __restrict__ stats,
                                float* __restrict__ h, int n64) {
  const int tid = blockIdx.x * blockDim.x + threadIdx.x;
  if (tid >= n64) return;
  const int c = tid & (HCH - 1);
  const float v = hnew[tid] * stats[128 + c] + stats[192 + c];
  h[tid] = v > 0.f ? v : 0.f;
}

__global__ void zero_f32_kernel(float* __restrict__ p, int n) {
  const int tid = blockIdx.x * blockDim.x + threadIdx.x;
  if (tid < n) p[tid] = 0.f;
}

// pool[b,c] += h[n,c]; cnt[b] += 1
__global__ void pool_kernel(const float* __restrict__ h,
                            const int* __restrict__ batch,
                            float* __restrict__ pool, float* __restrict__ cnt,
                            int N) {
  const int tid = blockIdx.x * blockDim.x + threadIdx.x;
  if (tid >= N * HCH) return;
  const int n = tid >> 6, c = tid & (HCH - 1);
  const int b = batch[n];
  atomicAdd(&pool[b * HCH + c], h[tid]);
  if (c == 0) atomicAdd(&cnt[b], 1.f);
}

// out[b,o] = (pool[b]/max(cnt,1)) @ W_cls + b_cls
__global__ void cls_kernel(const float* __restrict__ pool,
                           const float* __restrict__ cnt,
                           const float* __restrict__ Wc,
                           const float* __restrict__ bc,
                           float* __restrict__ out) {
  const int tid = blockIdx.x * blockDim.x + threadIdx.x;
  if (tid >= NGRAPH * OUTC) return;
  const int b = tid / OUTC, o = tid % OUTC;
  float cn = cnt[b]; cn = cn > 1.f ? cn : 1.f;
  const float inv = 1.f / cn;
  float acc = bc[o];
#pragma unroll
  for (int c = 0; c < HCH; ++c)
    acc += pool[b * HCH + c] * inv * Wc[c * OUTC + o];
  out[tid] = acc;
}

extern "C" void kernel_launch(void* const* d_in, const int* in_sizes, int n_in,
                              void* d_out, int out_size, void* d_ws, size_t ws_size,
                              hipStream_t stream) {
  const float* x      = (const float*)d_in[0];
  const int*   ei     = (const int*)d_in[1];
  const int*   batch  = (const int*)d_in[2];
  const float* W_enc  = (const float*)d_in[3];
  const float* b_enc  = (const float*)d_in[4];
  const float* Wl     = (const float*)d_in[5];
  const float* bl     = (const float*)d_in[6];
  const float* Wr     = (const float*)d_in[7];
  const float* br     = (const float*)d_in[8];
  const float* att    = (const float*)d_in[9];
  const float* bias   = (const float*)d_in[10];
  const float* gamma  = (const float*)d_in[11];
  const float* beta   = (const float*)d_in[12];
  const float* W_cls  = (const float*)d_in[13];
  const float* b_cls  = (const float*)d_in[14];
  float* out = (float*)d_out;

  const int N  = in_sizes[0] / 64;       // 100000
  const int E  = in_sizes[1] / 2;        // 1600000
  const int Ep = E + N;                  // + self loops

  // workspace layout (floats)
  float*    ws    = (float*)d_ws;
  float*    h     = ws;                                    // N*64
  float*    xl    = h    + (size_t)N * HCH;                // N*128
  float*    xr    = xl   + (size_t)N * CHF;                // N*128
  float*    hnew  = xr   + (size_t)N * CHF;                // N*64
  float*    esc   = hnew + (size_t)N * HCH;                // Ep*2
  unsigned* mOrd  = (unsigned*)(esc + (size_t)Ep * HEADS); // N*2
  float*    den   = (float*)mOrd + (size_t)N * HEADS;      // N*2
  float*    stats = den + (size_t)N * HEADS;               // 256
  float*    pool  = stats + 256;                           // 64*64
  float*    cnt   = pool + NGRAPH * HCH;                   // 64

  const int rowTiles = (N + 15) / 16;
  const int edgeBlocks = (Ep + 7) / 8;                     // 8 edges / 256-thr block
  const int initBlocks = (N * HCH + 255) / 256;

  // encoder: h = x @ W_enc + b_enc  (KO=64 -> 4 waves)
  gemm_wmma_kernel<64><<<rowTiles, 128, 0, stream>>>(x, W_enc, b_enc, h, N);

  for (int l = 0; l < 2; ++l) {
    const float* Wl_l = Wl + (size_t)l * HCH * CHF;
    const float* Wr_l = Wr + (size_t)l * HCH * CHF;
    // xl/xr transforms (KO=128 -> 8 waves)
    gemm_wmma_kernel<128><<<rowTiles, 256, 0, stream>>>(h, Wl_l, bl + l * CHF, xl, N);
    gemm_wmma_kernel<128><<<rowTiles, 256, 0, stream>>>(h, Wr_l, br + l * CHF, xr, N);

    layer_init_kernel<<<initBlocks, 256, 0, stream>>>(hnew, bias + l * HCH,
                                                      mOrd, den, stats, N);
    edge_score_kernel<<<edgeBlocks, 256, 0, stream>>>(xl, xr, att + l * CHF, ei,
                                                      esc, mOrd, E, Ep);
    edge_exp_kernel<<<(HEADS * Ep + 255) / 256, 256, 0, stream>>>(ei, mOrd, esc,
                                                                  den, E, Ep);
    edge_aggr_kernel<<<edgeBlocks, 256, 0, stream>>>(xl, ei, esc, den, hnew, E, Ep);

    bn_stats_kernel<<<(N + 1023) / 1024, dim3(64, 4), 0, stream>>>(hnew, stats, N);
    bn_finalize_kernel<<<1, 64, 0, stream>>>(stats, gamma + l * HCH,
                                             beta + l * HCH, N);
    bn_apply_kernel<<<initBlocks, 256, 0, stream>>>(hnew, stats, h, N * HCH);
  }

  zero_f32_kernel<<<(NGRAPH * (HCH + 1) + 255) / 256, 256, 0, stream>>>(
      pool, NGRAPH * (HCH + 1));
  pool_kernel<<<initBlocks, 256, 0, stream>>>(h, batch, pool, cnt, N);
  cls_kernel<<<(NGRAPH * OUTC + 255) / 256, 256, 0, stream>>>(pool, cnt, W_cls,
                                                              b_cls, out);
}
